// NodeModel_5935644803811
// MI455X (gfx1250) — matrix operand order repaired and checked by reading, not compiled, not run on hardware
//
#include <hip/hip_runtime.h>
#include <hip/hip_bf16.h>

// ---------------------------------------------------------------------------
// GraphNet node block on MI455X (gfx1250), bf16 WMMA pipeline:
//   GEMM1: relu(concat(x[row], edge_attr) @ W1 + b1)        -> h1 (bf16)
//   GEMM2: (h1 @ W2 + b2) scatter-added into sums[col]      (f32 atomics)
//   mean : agg = sums / max(cnt, 1)
//   GEMM3: relu(concat(x, agg) @ W3 + b3)                   -> h3 (bf16)
//   GEMM4: h3 @ W4 + b4                                     -> d_out (f32)
// Double-buffered LDS staging; AMODE-1 A-tiles use async global->LDS copies.
// ---------------------------------------------------------------------------

#define NNODES 50000
#define NEDGES 128000
#define FNODE  512
#define FEDGE  128
#define INSZ   640
#define HIDSZ  1280
#define IN2SZ  1792

typedef __bf16        v16bf __attribute__((ext_vector_type(16)));
typedef float         v8f   __attribute__((ext_vector_type(8)));
typedef unsigned int  v4u   __attribute__((ext_vector_type(4)));
typedef unsigned int  v2u   __attribute__((ext_vector_type(2)));
typedef int           v4i_t __attribute__((vector_size(16)));   // builtin param type

union FragBF { v16bf bf; v4u q[2]; };

__device__ __forceinline__ unsigned short f2bf(float f) {
    unsigned int u = __float_as_uint(f);
    u += 0x7FFFu + ((u >> 16) & 1u);          // round-to-nearest-even
    return (unsigned short)(u >> 16);
}
__device__ __forceinline__ unsigned int pack2bf(float a, float b) {
    return (unsigned int)f2bf(a) | ((unsigned int)f2bf(b) << 16);
}

// LDS pitches in ushort units: 40*2 = 80 bytes/row -> 16B-aligned fragment reads
#define APITCH 40
#define BPITCH 40

// ---- CDNA5 async global->LDS copy (ASYNCcnt-tracked) ----------------------
__device__ __forceinline__ void async_copy16(const void* gsrc, void* ldst) {
    unsigned long long g = (unsigned long long)(size_t)gsrc;
    unsigned int       l = (unsigned int)(size_t)ldst;   // low 32 bits = LDS offset
#if __has_builtin(__builtin_amdgcn_global_load_async_to_lds_b128)
    __builtin_amdgcn_global_load_async_to_lds_b128(
        (__attribute__((address_space(1))) v4i_t*)g,
        (__attribute__((address_space(3))) v4i_t*)l, 0, 0);
#else
    asm volatile("global_load_async_to_lds_b128 %0, %1, off"
                 :: "v"(l), "v"(g) : "memory");
#endif
}
__device__ __forceinline__ void wait_async0() {
#if __has_builtin(__builtin_amdgcn_s_wait_asynccnt)
    __builtin_amdgcn_s_wait_asynccnt(0);
#else
    asm volatile("s_wait_asynccnt 0" ::: "memory");
#endif
}

// AMODE: 0 = gather concat(x[rowIdx[m]], edge_attr[m])   (f32 -> bf16 inline)
//        1 = direct bf16 activation matrix Abf[M][K]     (async -> LDS)
//        2 = concat(x[m], agg[m])                        (f32 -> bf16 inline)
// EPI:   0 = relu -> bf16 store   1 = bias + atomic scatter-add   2 = f32 store
template <int AMODE, int EPI>
__global__ __launch_bounds__(256) void gemm_bf16_wmma(
    const float*          __restrict__ srcA0,   // x          (modes 0,2)
    const float*          __restrict__ srcA1,   // edge_attr (0) / agg (2)
    const unsigned short* __restrict__ Abf,     // mode 1 activations
    const unsigned short* __restrict__ Bbf,     // weights [K][N] bf16
    const float*          __restrict__ bias,    // [N]
    const int*            __restrict__ rowIdx,  // mode 0 gather rows
    const int*            __restrict__ colIdx,  // EPI 1 scatter rows
    unsigned short*       __restrict__ outBf,   // EPI 0
    float*                __restrict__ outF,    // EPI 1 (sums) / EPI 2
    int M, int N, int K)
{
    __shared__ __attribute__((aligned(16))) unsigned short lA[2][128 * APITCH];
    __shared__ __attribute__((aligned(16))) unsigned short lB[2][64 * BPITCH];
    __shared__ int cmap[128];

    const int tid  = threadIdx.x;
    const int lane = tid & 31;
    const int w    = tid >> 5;          // 8 waves
    const int wm0  = (w & 3) * 32;      // 4 waves along M
    const int wn0  = (w >> 2) * 32;     // 2 waves along N
    const int m0   = blockIdx.x * 128;
    const int n0   = blockIdx.y * 64;

    if (EPI == 1 && tid < 128) {
        int gm = m0 + tid;
        cmap[tid] = (gm < M) ? colIdx[gm] : -1;
    }

    // ---- hoist per-thread staging source pointers (loop-invariant) --------
    const unsigned short* aSrc16[2];    // AMODE 1: 2 tasks x 16B
    const float* aSrcX[4];              // AMODE 0/2: x-part base (+k0 in loop)
    const float* aSrcE[4];              //            edge/agg-part base
    int kgA[4];
    if (AMODE == 1) {
#pragma unroll
        for (int it = 0; it < 2; ++it) {
            int task = it * 256 + tid;          // 512 tasks of 8 bf16
            int r    = task >> 2;
            int kg   = (task & 3) * 8;
            int gm   = m0 + r; if (gm >= M) gm = M - 1;
            aSrc16[it] = Abf + (size_t)gm * K + kg;
        }
    } else {
#pragma unroll
        for (int it = 0; it < 4; ++it) {
            int task = it * 256 + tid;          // 1024 tasks of 4 f32
            int r    = task >> 3;
            int kg   = (task & 7) * 4;
            kgA[it]  = kg;
            if (AMODE == 0) {
                int gm = m0 + r;                 // NEDGES % 128 == 0
                int rr = rowIdx[gm];
                aSrcX[it] = srcA0 + (size_t)rr * FNODE + kg;
                aSrcE[it] = srcA1 + (size_t)gm * FEDGE + kg - FNODE;
            } else {
                int gm = m0 + r; if (gm >= M) gm = M - 1;
                aSrcX[it] = srcA0 + (size_t)gm * FNODE + kg;
                aSrcE[it] = srcA1 + (size_t)gm * HIDSZ + kg - FNODE;
            }
        }
    }
    const int kl = tid >> 3;                    // B staging: 0..31
    const int nl = (tid & 7) * 8;               //            0..56
    const unsigned short* bSrc = Bbf + (size_t)kl * N + (n0 + nl);

    // ---- tile staging into LDS buffer `buf` for K offset `k0s` ------------
    auto stage = [&](int buf, int k0s) {
        if (AMODE == 1) {
#pragma unroll
            for (int it = 0; it < 2; ++it) {
                int task = it * 256 + tid;
                int r    = task >> 2;
                int kg   = (task & 3) * 8;
                async_copy16(aSrc16[it] + k0s, &lA[buf][r * APITCH + kg]);
            }
        } else {
#pragma unroll
            for (int it = 0; it < 4; ++it) {
                int task = it * 256 + tid;
                int r    = task >> 3;
                int k    = k0s + kgA[it];
                const float* s = (k < FNODE) ? (aSrcX[it] + k0s)
                                             : (aSrcE[it] + k0s);
                float4 f = *(const float4*)s;
                v2u pk = { pack2bf(f.x, f.y), pack2bf(f.z, f.w) };
                *(v2u*)(&lA[buf][r * APITCH + kgA[it]]) = pk;
            }
        }
        {   // B tile, stored transposed (n-major) for contiguous-K fragments
            const unsigned short* p = bSrc + (size_t)k0s * N;
            __builtin_prefetch(p + (size_t)32 * N, 0, 1);   // global_prefetch
            v4u v = *(const v4u*)p;
            const unsigned short* sv = (const unsigned short*)&v;
#pragma unroll
            for (int j = 0; j < 8; ++j)
                lB[buf][(nl + j) * BPITCH + kl] = sv[j];
        }
    };

    v8f acc[2][2];
#pragma unroll
    for (int i = 0; i < 2; ++i)
#pragma unroll
        for (int j = 0; j < 2; ++j)
            acc[i][j] = (v8f){0.f, 0.f, 0.f, 0.f, 0.f, 0.f, 0.f, 0.f};

    const int mrow = wm0 + (lane & 15);
    const int akb  = (lane >> 4) * 8;           // A: K-half pairs per lane group
    const int ncol = wn0 + (lane & 15);
    const int bkb  = (lane >> 4) * 8;           // B: K-half pairs per lane group

    const int nsteps = K >> 5;
    stage(0, 0);                                 // prologue fill
    for (int s = 0; s < nsteps; ++s) {
        const int cur = s & 1;
        if (AMODE == 1) wait_async0();           // own async writes done
        __syncthreads();                         // publish buf `cur`
        if (s + 1 < nsteps) stage(cur ^ 1, (s + 1) << 5);

        FragBF a[2], b[2];
#pragma unroll
        for (int mf = 0; mf < 2; ++mf) {
            const unsigned short* base = &lA[cur][(mrow + 16 * mf) * APITCH];
            a[mf].q[0] = *(const v4u*)(base + akb);        // K akb..akb+7
            a[mf].q[1] = *(const v4u*)(base + akb + 16);   // K akb+16..akb+23
        }
#pragma unroll
        for (int nf = 0; nf < 2; ++nf) {
            const unsigned short* base = &lB[cur][(ncol + 16 * nf) * BPITCH + bkb];
            b[nf].q[0] = *(const v4u*)(base);
            b[nf].q[1] = *(const v4u*)(base + 16);
        }
#pragma unroll
        for (int mf = 0; mf < 2; ++mf)
#pragma unroll
            for (int nf = 0; nf < 2; ++nf)
                acc[mf][nf] = __builtin_amdgcn_wmma_f32_16x16x32_bf16(
                    false, a[mf].bf, false, b[nf].bf,
                    (short)0, acc[mf][nf], false, false);
    }

    // ---- epilogue ----
    const int ncol0 = n0 + wn0 + (lane & 15);
    float bv0 = bias[ncol0];
    float bv1 = bias[ncol0 + 16];
    const int lrow0 = wm0 + 8 * (lane >> 4);
#pragma unroll
    for (int mf = 0; mf < 2; ++mf) {
#pragma unroll
        for (int r = 0; r < 8; ++r) {
            int lrow = lrow0 + 16 * mf + r;
            int gm   = m0 + lrow;
            if (gm >= M) continue;
#pragma unroll
            for (int nf = 0; nf < 2; ++nf) {
                int gn  = ncol0 + 16 * nf;
                float v = acc[mf][nf][r] + (nf ? bv1 : bv0);
                if (EPI == 0) {
                    v = v > 0.f ? v : 0.f;
                    outBf[(size_t)gm * N + gn] = f2bf(v);
                } else if (EPI == 1) {
                    int dst = cmap[lrow];
                    if (dst >= 0)
                        atomicAdd(&outF[(size_t)dst * N + gn], v);
                } else {
                    outF[(size_t)gm * N + gn] = v;
                }
            }
        }
    }
}

__global__ void cvt_f32_bf16(const float* __restrict__ s,
                             unsigned short* __restrict__ d, int n) {
    int i = blockIdx.x * blockDim.x + threadIdx.x;
    if (i < n) d[i] = f2bf(s[i]);
}

__global__ void count_edges(const int* __restrict__ col,
                            float* __restrict__ cnt, int e_total) {
    int e = blockIdx.x * blockDim.x + threadIdx.x;
    if (e < e_total) atomicAdd(&cnt[col[e]], 1.0f);
}

__global__ void finalize_mean(float* __restrict__ sums,
                              const float* __restrict__ cnt,
                              int n_nodes, int hid) {
    size_t i = (size_t)blockIdx.x * blockDim.x + threadIdx.x;
    size_t total = (size_t)n_nodes * hid;
    if (i < total) {
        float c = cnt[i / hid];
        sums[i] = sums[i] / fmaxf(c, 1.0f);
    }
}

extern "C" void kernel_launch(void* const* d_in, const int* in_sizes, int n_in,
                              void* d_out, int out_size, void* d_ws, size_t ws_size,
                              hipStream_t stream) {
    (void)in_sizes; (void)n_in; (void)out_size; (void)ws_size;
    const float* x          = (const float*)d_in[0];
    const int*   edge_index = (const int*)d_in[1];      // [2][E]: row, col
    const float* edge_attr  = (const float*)d_in[2];
    const float* W1 = (const float*)d_in[3];  const float* b1 = (const float*)d_in[4];
    const float* W2 = (const float*)d_in[5];  const float* b2 = (const float*)d_in[6];
    const float* W3 = (const float*)d_in[7];  const float* b3 = (const float*)d_in[8];
    const float* W4 = (const float*)d_in[9];  const float* b4 = (const float*)d_in[10];
    const int* rowI = edge_index;
    const int* colI = edge_index + NEDGES;

    // workspace carve-up
    char* ws = (char*)d_ws;
    size_t off = 0;
    auto take = [&](size_t bytes) { char* p = ws + off; off += (bytes + 255) & ~(size_t)255; return p; };
    unsigned short* W1bf = (unsigned short*)take((size_t)INSZ  * HIDSZ * 2);
    unsigned short* W2bf = (unsigned short*)take((size_t)HIDSZ * HIDSZ * 2);
    unsigned short* W3bf = (unsigned short*)take((size_t)IN2SZ * INSZ  * 2);
    unsigned short* W4bf = (unsigned short*)take((size_t)INSZ  * FNODE * 2);
    unsigned short* h1bf = (unsigned short*)take((size_t)NEDGES * HIDSZ * 2);
    float*          sums = (float*)take((size_t)NNODES * HIDSZ * 4);
    float*          cnt  = (float*)take((size_t)NNODES * 4);
    unsigned short* h3bf = (unsigned short*)take((size_t)NNODES * INSZ * 2);

    (void)hipMemsetAsync(sums, 0, (size_t)NNODES * HIDSZ * 4, stream);
    (void)hipMemsetAsync(cnt,  0, (size_t)NNODES * 4, stream);

    // weight conversions f32 -> bf16
    {
        int n1 = INSZ * HIDSZ, n2 = HIDSZ * HIDSZ, n3 = IN2SZ * INSZ, n4 = INSZ * FNODE;
        cvt_f32_bf16<<<(n1 + 255) / 256, 256, 0, stream>>>(W1, W1bf, n1);
        cvt_f32_bf16<<<(n2 + 255) / 256, 256, 0, stream>>>(W2, W2bf, n2);
        cvt_f32_bf16<<<(n3 + 255) / 256, 256, 0, stream>>>(W3, W3bf, n3);
        cvt_f32_bf16<<<(n4 + 255) / 256, 256, 0, stream>>>(W4, W4bf, n4);
    }

    // GEMM1: [E,640]x[640,1280] relu -> h1 bf16
    {
        dim3 g(NEDGES / 128, HIDSZ / 64);
        gemm_bf16_wmma<0, 0><<<g, 256, 0, stream>>>(
            x, edge_attr, nullptr, W1bf, b1, rowI, nullptr,
            h1bf, nullptr, NEDGES, HIDSZ, INSZ);
    }
    // GEMM2: [E,1280]x[1280,1280] + b2, scatter-add into sums[col]
    {
        dim3 g(NEDGES / 128, HIDSZ / 64);
        gemm_bf16_wmma<1, 1><<<g, 256, 0, stream>>>(
            nullptr, nullptr, h1bf, W2bf, b2, nullptr, colI,
            nullptr, sums, NEDGES, HIDSZ, HIDSZ);
    }
    count_edges<<<(NEDGES + 255) / 256, 256, 0, stream>>>(colI, cnt, NEDGES);
    {
        size_t total = (size_t)NNODES * HIDSZ;
        finalize_mean<<<(unsigned)((total + 255) / 256), 256, 0, stream>>>(
            sums, cnt, NNODES, HIDSZ);
    }
    // GEMM3: [N,1792]x[1792,640] relu -> h3 bf16
    {
        dim3 g((NNODES + 127) / 128, INSZ / 64);
        gemm_bf16_wmma<2, 0><<<g, 256, 0, stream>>>(
            x, sums, nullptr, W3bf, b3, nullptr, nullptr,
            h3bf, nullptr, NNODES, INSZ, IN2SZ);
    }
    // GEMM4: [N,640]x[640,512] -> d_out f32
    {
        dim3 g((NNODES + 127) / 128, FNODE / 64);
        gemm_bf16_wmma<1, 2><<<g, 256, 0, stream>>>(
            nullptr, nullptr, h3bf, W4bf, b4, nullptr, nullptr,
            nullptr, (float*)d_out, NNODES, FNODE, INSZ);
    }
}
